// GATEncoder_22273700397755
// MI455X (gfx1250) — compile-verified, matrix operand order checked by reading, not run on hardware
//
#include <hip/hip_runtime.h>
#include <hip/hip_bf16.h>
#include <math.h>

// ---------------------------------------------------------------------------
// GAT encoder for MI455X (gfx1250):
//   - bf16 WMMA GEMMs (packed-fragment B, 4 N-tiles/wave)
//   - GAT GEMMs stage the shared A tile via the Tensor Data Mover
//     (tensor_load_to_lds, double-buffered LDS, s_wait_tensorcnt)
//   - attention softmax/aggregation via f32 atomics (bandwidth-bound part)
// ---------------------------------------------------------------------------

typedef __attribute__((ext_vector_type(16))) __bf16        v16bf;
typedef __attribute__((ext_vector_type(8)))  float         v8f;
typedef __attribute__((ext_vector_type(4)))  unsigned int  v4u;
typedef __attribute__((ext_vector_type(8)))  int           v8i;
typedef __attribute__((ext_vector_type(4)))  int           v4i;

union Frag {            // 16 bf16 = 32 bytes = two 16-byte vector loads
  v16bf v;
  v4u   q[2];
};

#define GN 10000      // nodes
#define GE 160000     // edges (w/o self loops)
#define GH 8          // heads
#define GC 128        // per-head dim
#define GD 1024       // concat dim = GH*GC
#define KPAD 32       // padded K for layer-1 GEMM

// -------------------- elementwise helpers --------------------

__global__ void fill_f32(float* __restrict__ p, float v, int n) {
  int i = blockIdx.x * blockDim.x + threadIdx.x;
  if (i < n) p[i] = v;
}

// dst[drows,dcols] (bf16) = src[srows,scols] (f32), zero padded (activations)
__global__ void pad_cvt_bf16(const float* __restrict__ src, __bf16* __restrict__ dst,
                             int srows, int scols, int drows, int dcols) {
  int i = blockIdx.x * blockDim.x + threadIdx.x;
  int total = drows * dcols;
  if (i >= total) return;
  int r = i / dcols, c = i - r * dcols;
  float v = (r < srows && c < scols) ? src[(size_t)r * scols + c] : 0.0f;
  dst[i] = (__bf16)v;
}

// out = bf16(gelu(in + bias[col]))   (exact erf gelu, matches approximate=False)
__global__ void bias_gelu_bf16(const float* __restrict__ in, const float* __restrict__ bias,
                               __bf16* __restrict__ out, int total, int cols) {
  int i = blockIdx.x * blockDim.x + threadIdx.x;
  if (i >= total) return;
  float v = in[i] + bias[i % cols];
  float g = 0.5f * v * (1.0f + erff(v * 0.70710678118654752f));
  out[i] = (__bf16)g;
}

// -------------------- B-matrix fragment packing --------------------
// Repack f32 B[Ksrc, Ncols] (row-major, zero-padded to Kpad rows) into the
// per-lane WMMA fragment layout so each lane's 16-element fragment is 32
// contiguous bytes:  dst[((kt*ntiles + nt)*32 + lane)*16 + elem]
//   lane half = lane>>4, col n = nt*16 + (lane&15)
//   k = kt*32 + 8*half + elem + (elem<8 ? 0 : 8)
__global__ void pack_b_wmma(const float* __restrict__ src, __bf16* __restrict__ dst,
                            int Ksrc, int Kpad, int Ncols) {
  long long total = (long long)Kpad * Ncols;
  long long i = (long long)blockIdx.x * blockDim.x + threadIdx.x;
  if (i >= total) return;
  int elem = (int)(i & 15);
  long long rest = i >> 4;
  int lane = (int)(rest & 31); rest >>= 5;
  int ntiles = Ncols >> 4;
  int nt = (int)(rest % ntiles);
  int kt = (int)(rest / ntiles);
  int half = lane >> 4;
  int n = nt * 16 + (lane & 15);
  int k = kt * 32 + 8 * half + elem + (elem < 8 ? 0 : 8);
  float v = (k < Ksrc) ? src[(size_t)k * Ncols + n] : 0.0f;
  dst[i] = (__bf16)v;
}

// -------------------- TDM helper --------------------
// Issue a Tensor Data Mover load of a 2-D bf16 tile (rows x cols elements,
// row stride in elements) from global memory into LDS at byte offset lds_off.
// D# built per CDNA5 ISA (group0/group1; trailing groups
// zero for a <=2-D tensor).  clang-23 6-arg builtin form.
__device__ __forceinline__ void tdm_load_tile_bf16(const void* gsrc, unsigned int lds_off,
                                                   int rows, int cols, int rowstride) {
  unsigned long long ga = (unsigned long long)gsrc;
  v4u g0;
  g0[0] = 1u;                                          // count=1, user mode
  g0[1] = lds_off;                                     // lds_addr (bytes)
  g0[2] = (unsigned int)(ga & 0xffffffffu);            // global_addr[31:0]
  g0[3] = (unsigned int)((ga >> 32) & 0x1ffffffu)      // global_addr[56:32]
          | 0x80000000u;                               // type=2 ("image")
  unsigned int td0 = (unsigned int)cols;               // tensor_dim0 (elems)
  unsigned int td1 = (unsigned int)rows;               // tensor_dim1
  unsigned long long st0 = (unsigned long long)rowstride;
  v8i g1;
  g1[0] = (int)(1u << 16);                             // data_size=1 (2 bytes)
  g1[1] = (int)((td0 & 0xffffu) << 16);                // tensor_dim0[15:0]
  g1[2] = (int)((td0 >> 16) | ((td1 & 0xffffu) << 16));// td0 hi | td1 lo
  g1[3] = (int)((td1 >> 16) | ((unsigned int)cols << 16)); // td1 hi | tile_dim0
  g1[4] = (int)(rows & 0xffffu);                       // tile_dim1, tile_dim2=0
  g1[5] = (int)(st0 & 0xffffffffu);                    // tensor_dim0_stride lo
  g1[6] = (int)((st0 >> 32) & 0xffffu);                // stride hi | dim1_stride=0
  g1[7] = 0;
  v4i z4 = {0, 0, 0, 0};
  v8i z8 = {0, 0, 0, 0, 0, 0, 0, 0};
  __builtin_amdgcn_tensor_load_to_lds(g0, g1, z4, z4, z8, 0);
}

// -------------------- WMMA GEMM (register path) --------------------
// C[M,Ncols] = A[M,K] @ B[K,Ncols] (+bias), optional bf16 copy of C.
// A: plain row-major bf16 (per-lane fragment = two contiguous 16B chunks).
// B: packed fragment layout from pack_b_wmma (32B contiguous per lane).
template <int NT>
__global__ void wmma_gemm_bf16(const __bf16* __restrict__ A, const __bf16* __restrict__ Bp,
                               float* __restrict__ C, __bf16* __restrict__ Cbf,
                               const float* __restrict__ bias,
                               int M, int Ncols, int K) {
  const int wavesPerBlock = blockDim.x >> 5;
  int wid  = blockIdx.x * wavesPerBlock + (threadIdx.x >> 5);
  int lane = threadIdx.x & 31;
  int ntiles  = Ncols >> 4;
  int ngroups = ntiles / NT;
  int mtiles  = M >> 4;
  if (wid >= mtiles * ngroups) return;
  int mt = wid / ngroups, ng = wid - mt * ngroups;
  int nt0  = ng * NT;
  int row0 = mt << 4;

  int half = lane >> 4;
  int sub  = lane & 15;
  int rA   = row0 + sub;

  v8f acc[NT];
#pragma unroll
  for (int t = 0; t < NT; ++t) acc[t] = (v8f){};

  const int ktiles = K >> 5;
  for (int kt = 0; kt < ktiles; ++kt) {
    const __bf16* Arow = A + (size_t)rA * K + kt * 32 + 8 * half;
    Frag a;
    a.q[0] = *(const v4u*)(Arow);
    a.q[1] = *(const v4u*)(Arow + 16);
#pragma unroll
    for (int t = 0; t < NT; ++t) {
      const __bf16* bp = Bp + ((size_t)(kt * ntiles + nt0 + t) * 32 + lane) * 16;
      Frag b;
      b.q[0] = *(const v4u*)(bp);
      b.q[1] = *(const v4u*)(bp + 8);
      acc[t] = __builtin_amdgcn_wmma_f32_16x16x32_bf16(false, a.v, false, b.v,
                                                       (short)0, acc[t], false, false);
    }
  }

#pragma unroll
  for (int t = 0; t < NT; ++t) {
    int colC = (nt0 + t) * 16 + sub;
    float bv = bias ? bias[colC] : 0.0f;
#pragma unroll
    for (int r = 0; r < 8; ++r) {
      int rowC = row0 + r + 8 * half;
      float v = acc[t][r] + bv;
      size_t idx = (size_t)rowC * Ncols + colC;
      C[idx] = v;
      if (Cbf) Cbf[idx] = (__bf16)v;
    }
  }
}

// -------------------- WMMA GEMM (TDM-staged A path, GAT linears) --------------------
// All 8 waves of a block share one M-tile (requires ngroups % wavesPerBlock == 0,
// grid exact).  Wave 0 DMAs the 16x32 bf16 A tile for step kt+1 into the LDS
// back buffer with the Tensor Data Mover while all waves run WMMAs on the
// front buffer; sync = s_wait_tensorcnt 0 (issuer) + workgroup barrier.
template <int NT>
__global__ void wmma_gemm_bf16_tdm(const __bf16* __restrict__ A,
                                   const __bf16* __restrict__ Bp,
                                   float* __restrict__ C,
                                   int M, int Ncols, int K) {
  __shared__ __align__(128) __bf16 asmem[2][16 * 32];   // double-buffered A tile

  const int wavesPerBlock = blockDim.x >> 5;
  int wid  = blockIdx.x * wavesPerBlock + (threadIdx.x >> 5);
  int lane = threadIdx.x & 31;
  int ntiles  = Ncols >> 4;
  int ngroups = ntiles / NT;
  int mt = wid / ngroups, ng = wid - mt * ngroups;
  int nt0  = ng * NT;
  int row0 = mt << 4;

  int half = lane >> 4;
  int sub  = lane & 15;

  v8f acc[NT];
#pragma unroll
  for (int t = 0; t < NT; ++t) acc[t] = (v8f){};

  const int ktiles = K >> 5;
  const __bf16* Abase = A + (size_t)row0 * K;           // tile row base
  const bool issuer = (threadIdx.x == 0);               // one TDM per block

  if (issuer)
    tdm_load_tile_bf16(Abase, (unsigned int)(unsigned long long)&asmem[0][0],
                       16, 32, K);

  for (int kt = 0; kt < ktiles; ++kt) {
    int cur = kt & 1;
    if (issuer) __builtin_amdgcn_s_wait_tensorcnt(0);   // front buffer landed
    __syncthreads();                                     // visible to all waves
    if (issuer && (kt + 1) < ktiles)                     // prefetch next tile
      tdm_load_tile_bf16(Abase + (kt + 1) * 32,
                         (unsigned int)(unsigned long long)&asmem[cur ^ 1][0],
                         16, 32, K);

    // A fragment from LDS: row sub (stride 32 elems), chunks at 8h / 8h+16
    const __bf16* ab = &asmem[cur][0] + sub * 32 + 8 * half;
    Frag a;
    a.q[0] = *(const v4u*)(ab);
    a.q[1] = *(const v4u*)(ab + 16);

#pragma unroll
    for (int t = 0; t < NT; ++t) {
      const __bf16* bp = Bp + ((size_t)(kt * ntiles + nt0 + t) * 32 + lane) * 16;
      Frag b;
      b.q[0] = *(const v4u*)(bp);
      b.q[1] = *(const v4u*)(bp + 8);
      acc[t] = __builtin_amdgcn_wmma_f32_16x16x32_bf16(false, a.v, false, b.v,
                                                       (short)0, acc[t], false, false);
    }
  }

#pragma unroll
  for (int t = 0; t < NT; ++t) {
    int colC = (nt0 + t) * 16 + sub;
#pragma unroll
    for (int r = 0; r < 8; ++r) {
      int rowC = row0 + r + 8 * half;
      C[(size_t)rowC * Ncols + colC] = acc[t][r];
    }
  }
}

// -------------------- attention pieces --------------------

// es[n,h] = sum_c h[n,h,c]*a_s[h,c] ; ed likewise.  One thread per (n,h).
__global__ void attn_coeff(const float* __restrict__ h, const float* __restrict__ a_s,
                           const float* __restrict__ a_d,
                           float* __restrict__ es, float* __restrict__ ed) {
  int i = blockIdx.x * blockDim.x + threadIdx.x;   // i = n*GH + hh
  if (i >= GN * GH) return;
  int hh = i & (GH - 1);
  const float* row = h + (size_t)i * GC;
  const float* as  = a_s + hh * GC;
  const float* ad  = a_d + hh * GC;
  float s = 0.f, d = 0.f;
#pragma unroll 4
  for (int c = 0; c < GC; ++c) {
    float v = row[c];
    s += v * as[c];
    d += v * ad[c];
  }
  es[i] = s;
  ed[i] = d;
}

__device__ __forceinline__ void get_edge(const int* __restrict__ ei, int e, int& s, int& d) {
  if (e < GE) { s = ei[e]; d = ei[GE + e]; }
  else        { s = d = e - GE; }               // self loops appended
}

__device__ __forceinline__ void atomicMaxF32(float* addr, float val) {
  if (val >= 0.0f) atomicMax((int*)addr, __float_as_int(val));
  else             atomicMin((unsigned int*)addr, __float_as_uint(val));
}

// segment max of leaky-relu logits over dst
__global__ void edge_max(const int* __restrict__ ei, const float* __restrict__ es,
                         const float* __restrict__ ed, float* __restrict__ m) {
  int e = blockIdx.x * blockDim.x + threadIdx.x;
  if (e >= GE + GN) return;
  int s, d;
  get_edge(ei, e, s, d);
#pragma unroll
  for (int hh = 0; hh < GH; ++hh) {
    float v = es[s * GH + hh] + ed[d * GH + hh];
    v = v > 0.0f ? v : 0.2f * v;                 // LeakyReLU(0.2)
    atomicMaxF32(&m[d * GH + hh], v);
  }
}

// segment sum of exp(logit - max)
__global__ void edge_sum(const int* __restrict__ ei, const float* __restrict__ es,
                         const float* __restrict__ ed, const float* __restrict__ m,
                         float* __restrict__ z) {
  int e = blockIdx.x * blockDim.x + threadIdx.x;
  if (e >= GE + GN) return;
  int s, d;
  get_edge(ei, e, s, d);
#pragma unroll
  for (int hh = 0; hh < GH; ++hh) {
    float v = es[s * GH + hh] + ed[d * GH + hh];
    v = v > 0.0f ? v : 0.2f * v;
    atomicAdd(&z[d * GH + hh], expf(v - m[d * GH + hh]));
  }
}

// agg[dst,h,:] += alpha * h[src,h,:]   — one wave32 per (edge, head),
// each lane handles 4 channels (32 lanes * 4 = 128 = GC).
__global__ void edge_aggregate(const int* __restrict__ ei, const float* __restrict__ h,
                               const float* __restrict__ es, const float* __restrict__ ed,
                               const float* __restrict__ m, const float* __restrict__ z,
                               float* __restrict__ agg) {
  int wid  = blockIdx.x * (blockDim.x >> 5) + (threadIdx.x >> 5);
  int lane = threadIdx.x & 31;
  if (wid >= (GE + GN) * GH) return;
  int e  = wid >> 3;
  int hh = wid & (GH - 1);
  int s, d;
  get_edge(ei, e, s, d);

  const float* srow = h + (size_t)s * GD + hh * GC + lane * 4;
  __builtin_prefetch(srow, 0, 0);                // global_prefetch_b8

  float v = es[s * GH + hh] + ed[d * GH + hh];
  v = v > 0.0f ? v : 0.2f * v;
  float alpha = expf(v - m[d * GH + hh]) / z[d * GH + hh];

  float4 hv = *(const float4*)srow;
  float* dst = agg + (size_t)d * GD + hh * GC + lane * 4;
  atomicAdd(dst + 0, hv.x * alpha);
  atomicAdd(dst + 1, hv.y * alpha);
  atomicAdd(dst + 2, hv.z * alpha);
  atomicAdd(dst + 3, hv.w * alpha);
}

// -------------------- host orchestration --------------------

extern "C" void kernel_launch(void* const* d_in, const int* in_sizes, int n_in,
                              void* d_out, int out_size, void* d_ws, size_t ws_size,
                              hipStream_t stream) {
  const float* x     = (const float*)d_in[0];
  const int*   ei    = (const int*)d_in[1];
  const float* w1    = (const float*)d_in[2];
  const float* asrc1 = (const float*)d_in[3];
  const float* adst1 = (const float*)d_in[4];
  const float* bias1 = (const float*)d_in[5];
  const float* w2    = (const float*)d_in[6];
  const float* asrc2 = (const float*)d_in[7];
  const float* adst2 = (const float*)d_in[8];
  const float* bias2 = (const float*)d_in[9];
  const float* w3    = (const float*)d_in[10];
  const float* asrc3 = (const float*)d_in[11];
  const float* adst3 = (const float*)d_in[12];
  const float* bias3 = (const float*)d_in[13];
  const float* rw1   = (const float*)d_in[14];
  const float* rb1   = (const float*)d_in[15];
  const float* rw2   = (const float*)d_in[16];
  const float* rb2   = (const float*)d_in[17];
  const float* lw    = (const float*)d_in[18];
  const float* lb    = (const float*)d_in[19];
  float* out = (float*)d_out;

  // ---- carve workspace ----
  char* p = (char*)d_ws;
  auto carve = [&](size_t bytes) -> char* {
    char* r = p;
    p += (bytes + 255) & ~(size_t)255;
    return r;
  };
  float*  hbuf   = (float*)carve((size_t)GN * GD * 4);   // GAT linear output
  float*  aggbuf = (float*)carve((size_t)GN * GD * 4);   // attention aggregation
  __bf16* actbf  = (__bf16*)carve((size_t)GN * GD * 2);  // gelu'd GAT output (bf16)
  float*  x128f  = (float*)carve((size_t)GN * GC * 4);   // projection output f32
  __bf16* x128bf = (__bf16*)carve((size_t)GN * GC * 2);  // projection output bf16
  __bf16* xpadbf = (__bf16*)carve((size_t)GN * KPAD * 2);
  __bf16* w1p    = (__bf16*)carve((size_t)KPAD * GD * 2);   // packed fragments
  __bf16* w2p    = (__bf16*)carve((size_t)GC * GD * 2);
  __bf16* w3p    = (__bf16*)carve((size_t)GC * GD * 2);
  __bf16* rw1p   = (__bf16*)carve((size_t)GD * GC * 2);
  __bf16* rw2p   = (__bf16*)carve((size_t)GD * GC * 2);
  __bf16* lwp    = (__bf16*)carve((size_t)GD * GC * 2);
  float*  es     = (float*)carve((size_t)GN * GH * 4);
  float*  ed     = (float*)carve((size_t)GN * GH * 4);
  float*  mbuf   = (float*)carve((size_t)GN * GH * 4);
  float*  zbuf   = (float*)carve((size_t)GN * GH * 4);
  (void)ws_size; (void)in_sizes; (void)n_in; (void)out_size;

  const int B = 256;
  auto blocks = [](long long n, int b) { return (int)((n + b - 1) / b); };

  // ---- one-time (per launch): pad x, pack all weights into fragment layout ----
  pad_cvt_bf16<<<blocks((long long)GN * KPAD, B), B, 0, stream>>>(x, xpadbf, GN, 4, GN, KPAD);
  pack_b_wmma<<<blocks((long long)KPAD * GD, B), B, 0, stream>>>(w1, w1p, 4, KPAD, GD);
  pack_b_wmma<<<blocks((long long)GC * GD, B), B, 0, stream>>>(w2, w2p, GC, GC, GD);
  pack_b_wmma<<<blocks((long long)GC * GD, B), B, 0, stream>>>(w3, w3p, GC, GC, GD);
  pack_b_wmma<<<blocks((long long)GD * GC, B), B, 0, stream>>>(rw1, rw1p, GD, GD, GC);
  pack_b_wmma<<<blocks((long long)GD * GC, B), B, 0, stream>>>(rw2, rw2p, GD, GD, GC);
  pack_b_wmma<<<blocks((long long)GD * GC, B), B, 0, stream>>>(lw, lwp, GD, GD, GC);

  const int ET = GE + GN;  // edges incl. self loops
  const int NT = 4;        // n-tiles per wave in the GEMM

  auto run_gat = [&](const __bf16* inbf, int K, const __bf16* wp,
                     const float* a_s, const float* a_d, const float* b_) {
    // h = in @ w via TDM-staged GEMM (ngroups = 64/NT = 16, exact grid:
    // all 8 waves of a block share one M-tile)
    int waves = (GN / 16) * (GD / 16) / NT;
    wmma_gemm_bf16_tdm<NT><<<blocks(waves, 8), B, 0, stream>>>(inbf, wp, hbuf,
                                                               GN, GD, K);
    attn_coeff<<<blocks((long long)GN * GH, B), B, 0, stream>>>(hbuf, a_s, a_d, es, ed);
    fill_f32<<<blocks((long long)GN * GH, B), B, 0, stream>>>(mbuf, -3.402823466e38f, GN * GH);
    fill_f32<<<blocks((long long)GN * GH, B), B, 0, stream>>>(zbuf, 0.0f, GN * GH);
    fill_f32<<<blocks((long long)GN * GD, B), B, 0, stream>>>(aggbuf, 0.0f, GN * GD);
    edge_max<<<blocks(ET, B), B, 0, stream>>>(ei, es, ed, mbuf);
    edge_sum<<<blocks(ET, B), B, 0, stream>>>(ei, es, ed, mbuf, zbuf);
    // one wave per (edge, head): ET blocks of 8 waves
    edge_aggregate<<<ET, B, 0, stream>>>(ei, hbuf, es, ed, mbuf, zbuf, aggbuf);
    // act = bf16(gelu(agg + bias))
    bias_gelu_bf16<<<blocks((long long)GN * GD, B), B, 0, stream>>>(aggbuf, b_, actbf,
                                                                    GN * GD, GD);
  };

  int pwaves = (GN / 16) * (GC / 16) / NT;

  // ---- layer 1 ----
  run_gat(xpadbf, KPAD, w1p, asrc1, adst1, bias1);
  wmma_gemm_bf16<NT><<<blocks(pwaves, 8), B, 0, stream>>>(actbf, rw1p, x128f, x128bf,
                                                          rb1, GN, GC, GD);
  // ---- layer 2 ----
  run_gat(x128bf, GC, w2p, asrc2, adst2, bias2);
  wmma_gemm_bf16<NT><<<blocks(pwaves, 8), B, 0, stream>>>(actbf, rw2p, x128f, x128bf,
                                                          rb2, GN, GC, GD);
  // ---- layer 3 ----
  run_gat(x128bf, GC, w3p, asrc3, adst3, bias3);
  // final projection straight into d_out (f32)
  wmma_gemm_bf16<NT><<<blocks(pwaves, 8), B, 0, stream>>>(actbf, lwp, out, nullptr,
                                                          lb, GN, GC, GD);
}
